// HS04Model_66984309948717
// MI455X (gfx1250) — matrix operand order repaired and checked by reading, not compiled, not run on hardware
//
#include <hip/hip_runtime.h>

typedef __attribute__((ext_vector_type(16))) __bf16 v16bf;
typedef __attribute__((ext_vector_type(8)))  __bf16 v8bf;
typedef __attribute__((ext_vector_type(8)))  float  v8f;
typedef __attribute__((ext_vector_type(4)))  float  v4f;

// Global (address_space(1)) pointer types so weight loads lower to
// 16-bit global loads (LOADcnt only) instead of flat loads (LOADcnt+DScnt,
// ties up the LDS path we need for A-fragments).
typedef const __bf16 __attribute__((address_space(1)))* gbfp;
typedef const v16bf  __attribute__((address_space(1)))* gv16p;

#define T_STEPS 20
#define BATCH   16384
#define P_DIM   256
#define H_DIM   512
#define S_DIM   512
#define C_DIM   128
#define BT      32              // batch rows per block
#define NTHREADS 512            // 16 waves -> 4 waves/SIMD -> 256 VGPR budget
#define LDH     520             // padded LDS stride for 512-wide activations
#define LDC     136             // padded LDS stride for 128-wide activations
#define SMEM_BYTES ((BT*LDH*2 + BT*LDC) * 2)

#define WMMA_BF16(A,B,C) \
  __builtin_amdgcn_wmma_f32_16x16x32_bf16(false, (A), false, (B), (short)0, (C), false, false)

__device__ __forceinline__ float sigmoid_f(float x) {
  float e = __builtin_amdgcn_exp2f(x * -1.44269504f);
  return __builtin_amdgcn_rcpf(1.0f + e);
}

// A-fragment (16x32 bf16) from LDS activations: lane ln=row, kh selects K-half.
__device__ __forceinline__ v16bf ldsA(const __bf16* base, int ln, int kh, int k0, int ld) {
  const __bf16* p = base + ln * ld + k0 + kh * 8;
  v8bf lo = *(const v8bf*)p;          // K = k0 + kh*8 + 0..7
  v8bf hi = *(const v8bf*)(p + 16);   // K = k0 + 16 + kh*8 + 0..7
  return __builtin_shufflevector(lo, hi, 0,1,2,3,4,5,6,7,8,9,10,11,12,13,14,15);
}

// B-fragment (32x16 bf16) from a row-major KxN bf16 weight matrix in global.
__device__ __forceinline__ v16bf gW(unsigned long long W, int k0, int ln, int kh, int n0, int N) {
  gbfp base = (gbfp)W + (size_t)(k0 + kh * 8) * N + n0 + ln;
  v16bf r;
#pragma unroll
  for (int s = 0; s < 8; ++s) {
    r[s]     = base[(size_t)s * N];
    r[8 + s] = base[(size_t)(16 + s) * N];
  }
  return r;
}

// A-fragment (16x32) for x_t, loaded f32 from global and converted to bf16.
__device__ __forceinline__ v16bf ldX(const float* x, int ln, int kh, int kc) {
  const float* p = x + ln * P_DIM + kc * 32 + kh * 8;
  v4f a0 = *(const v4f*)(p);
  v4f a1 = *(const v4f*)(p + 4);
  v4f b0 = *(const v4f*)(p + 16);
  v4f b1 = *(const v4f*)(p + 20);
  v16bf r;
#pragma unroll
  for (int i = 0; i < 4; ++i) {
    r[i]      = (__bf16)a0[i];
    r[4 + i]  = (__bf16)a1[i];
    r[8 + i]  = (__bf16)b0[i];
    r[12 + i] = (__bf16)b1[i];
  }
  return r;
}

__global__ void cvt_f32_bf16(const float* __restrict__ s, __bf16* __restrict__ d, int n) {
  int i = blockIdx.x * blockDim.x + threadIdx.x;
  if (i < n) d[i] = (__bf16)s[i];
}

__global__ __launch_bounds__(NTHREADS)
void hs04_kernel(const float* __restrict__ inputs,
                 const __bf16* __restrict__ w_ph, const __bf16* __restrict__ w_hs,
                 const __bf16* __restrict__ w_ss, const __bf16* __restrict__ w_sc,
                 const __bf16* __restrict__ w_cs,
                 const float* __restrict__ b_h, const float* __restrict__ b_s,
                 const float* __restrict__ b_c, float* __restrict__ out) {
  extern __shared__ __align__(16) char smem[];
  __bf16* aH = (__bf16*)smem;           // BT x 512 (stride LDH)
  __bf16* aS = aH + BT * LDH;           // BT x 512 (stride LDH)
  __bf16* aC = aS + BT * LDH;           // BT x 128 (stride LDC)

  const int tid  = threadIdx.x;
  const int wave = tid >> 5;
  const int lane = tid & 31;
  const int ln   = lane & 15;           // row (A) / column (B,C,D)
  const int kh   = lane >> 4;           // K-half (A,B) / row-half (C,D)
  const int mt   = wave >> 3;           // 0..1: 16-row batch subtile
  const int wq   = wave & 7;            // 0..7: column group
  const int b0   = blockIdx.x * BT;

  // Initialize activations to sigmoid(0)=0.5
  const int totalLds = BT * LDH * 2 + BT * LDC;
  for (int i = tid; i < totalLds; i += NTHREADS) ((__bf16*)smem)[i] = (__bf16)0.5f;

  // Per-lane bias (bias depends only on the output column)
  float biasH[4], biasS[4];
#pragma unroll
  for (int j = 0; j < 4; ++j) {
    int n = (wq * 4 + j) * 16 + ln;
    biasH[j] = b_h[n];
    biasS[j] = b_s[n];
  }
  float biasC = b_c[wq * 16 + ln];

  // Persistent leaky-integrator states as f32 accumulator fragments
  v8f inH[4] = {}, inS[4] = {};
  v8f inC = {};

  __syncthreads();

  const __bf16* aHrow = aH + mt * 16 * LDH;
  const __bf16* aSrow = aS + mt * 16 * LDH;
  const __bf16* aCrow = aC + mt * 16 * LDC;

  // Weight pointers as mutable 64-bit values so an empty asm can make them
  // loop-variant (defeats LICM hoisting+spilling of weight fragments).
  unsigned long long u_ph = (unsigned long long)w_ph;
  unsigned long long u_hs = (unsigned long long)w_hs;
  unsigned long long u_ss = (unsigned long long)w_ss;
  unsigned long long u_sc = (unsigned long long)w_sc;
  unsigned long long u_cs = (unsigned long long)w_cs;

  for (int t = 0; t < T_STEPS; ++t) {
    // Optimization barrier: weight base addresses appear redefined each step,
    // so weight loads cannot be hoisted out of the t-loop.
    asm volatile("" : "+s"(u_ph), "+s"(u_hs), "+s"(u_ss), "+s"(u_sc), "+s"(u_cs));

    const float* xblk = inputs + ((size_t)t * BATCH + b0 + mt * 16) * P_DIM;

    // ---- hps = x_t @ w_hps_ph : K = 256 ----
    v8f mmH[4] = {};
#pragma unroll
    for (int kc = 0; kc < P_DIM / 32; ++kc) {
      v16bf a = ldX(xblk, ln, kh, kc);
#pragma unroll
      for (int j = 0; j < 4; ++j)
        mmH[j] = WMMA_BF16(a, gW(u_ph, kc * 32, ln, kh, (wq * 4 + j) * 16, H_DIM), mmH[j]);
    }
    // fold hps integrator (registers only; frees mmH)
#pragma unroll
    for (int j = 0; j < 4; ++j)
#pragma unroll
      for (int v = 0; v < 8; ++v)
        inH[j][v] = 0.25f * (mmH[j][v] + biasH[j]) + 0.75f * inH[j][v];

    // ---- s += a_s @ w_ss ; css = a_s @ w_sc : shared A from aS, K = 512 ----
    v8f mmS[4] = {};
    v8f mmC = {};
#pragma unroll
    for (int kc = 0; kc < S_DIM / 32; ++kc) {
      v16bf a = ldsA(aSrow, ln, kh, kc * 32, LDH);
#pragma unroll
      for (int j = 0; j < 4; ++j)
        mmS[j] = WMMA_BF16(a, gW(u_ss, kc * 32, ln, kh, (wq * 4 + j) * 16, S_DIM), mmS[j]);
      mmC = WMMA_BF16(a, gW(u_sc, kc * 32, ln, kh, wq * 16, C_DIM), mmC);
    }
#pragma unroll
    for (int v = 0; v < 8; ++v)
      inC[v] = 0.25f * (mmC[v] + biasC) + 0.75f * inC[v];

    // ---- s += a_hps @ w_hps_hs : K = 512 ----
#pragma unroll
    for (int kc = 0; kc < H_DIM / 32; ++kc) {
      v16bf a = ldsA(aHrow, ln, kh, kc * 32, LDH);
#pragma unroll
      for (int j = 0; j < 4; ++j)
        mmS[j] = WMMA_BF16(a, gW(u_hs, kc * 32, ln, kh, (wq * 4 + j) * 16, S_DIM), mmS[j]);
    }
    // ---- s += a_css @ w_cs : K = 128 ----
#pragma unroll
    for (int kc = 0; kc < C_DIM / 32; ++kc) {
      v16bf a = ldsA(aCrow, ln, kh, kc * 32, LDC);
#pragma unroll
      for (int j = 0; j < 4; ++j)
        mmS[j] = WMMA_BF16(a, gW(u_cs, kc * 32, ln, kh, (wq * 4 + j) * 16, S_DIM), mmS[j]);
    }
#pragma unroll
    for (int j = 0; j < 4; ++j)
#pragma unroll
      for (int v = 0; v < 8; ++v)
        inS[j][v] = 0.25f * (mmS[j][v] + biasS[j]) + 0.75f * inS[j][v];

    __syncthreads();  // all waves done READING old activations

    // ---- write new activations (sigmoid of integrator states) ----
#pragma unroll
    for (int j = 0; j < 4; ++j) {
      int n = (wq * 4 + j) * 16 + ln;
#pragma unroll
      for (int v = 0; v < 8; ++v) {
        int r = mt * 16 + kh * 8 + v;
        aH[r * LDH + n] = (__bf16)sigmoid_f(inH[j][v]);
        float as = sigmoid_f(inS[j][v]);
        aS[r * LDH + n] = (__bf16)as;
        if (t >= T_STEPS - 4)
          out[((size_t)(t - (T_STEPS - 4)) * BATCH + b0 + r) * S_DIM + n] = as;
      }
    }
    {
      int n = wq * 16 + ln;
#pragma unroll
      for (int v = 0; v < 8; ++v) {
        int r = mt * 16 + kh * 8 + v;
        aC[r * LDC + n] = (__bf16)sigmoid_f(inC[v]);
      }
    }
    __syncthreads();  // new activations visible before next step reads
  }
}

extern "C" void kernel_launch(void* const* d_in, const int* in_sizes, int n_in,
                              void* d_out, int out_size, void* d_ws, size_t ws_size,
                              hipStream_t stream) {
  const float* inputs = (const float*)d_in[0];
  const float* wph_f  = (const float*)d_in[1];
  const float* whs_f  = (const float*)d_in[2];
  const float* wss_f  = (const float*)d_in[3];
  const float* wsc_f  = (const float*)d_in[4];
  const float* wcs_f  = (const float*)d_in[5];
  const float* b_h    = (const float*)d_in[6];
  const float* b_s    = (const float*)d_in[7];
  const float* b_c    = (const float*)d_in[8];

  // bf16 weight staging in workspace (1.5 MB, L2-resident)
  __bf16* w_ph = (__bf16*)d_ws;                         // 256*512
  __bf16* w_hs = w_ph + P_DIM * H_DIM;                  // 512*512
  __bf16* w_ss = w_hs + H_DIM * S_DIM;                  // 512*512
  __bf16* w_sc = w_ss + S_DIM * S_DIM;                  // 512*128
  __bf16* w_cs = w_sc + S_DIM * C_DIM;                  // 128*512

  cvt_f32_bf16<<<(P_DIM * H_DIM + 255) / 256, 256, 0, stream>>>(wph_f, w_ph, P_DIM * H_DIM);
  cvt_f32_bf16<<<(H_DIM * S_DIM + 255) / 256, 256, 0, stream>>>(whs_f, w_hs, H_DIM * S_DIM);
  cvt_f32_bf16<<<(S_DIM * S_DIM + 255) / 256, 256, 0, stream>>>(wss_f, w_ss, S_DIM * S_DIM);
  cvt_f32_bf16<<<(S_DIM * C_DIM + 255) / 256, 256, 0, stream>>>(wsc_f, w_sc, S_DIM * C_DIM);
  cvt_f32_bf16<<<(C_DIM * S_DIM + 255) / 256, 256, 0, stream>>>(wcs_f, w_cs, C_DIM * S_DIM);

  hipFuncSetAttribute(reinterpret_cast<const void*>(hs04_kernel),
                      hipFuncAttributeMaxDynamicSharedMemorySize, SMEM_BYTES);

  hs04_kernel<<<BATCH / BT, NTHREADS, SMEM_BYTES, stream>>>(
      inputs, w_ph, w_hs, w_ss, w_sc, w_cs, b_h, b_s, b_c, (float*)d_out);
}
